// SparseResidualBlock_67989332296241
// MI455X (gfx1250) — matrix operand order repaired and checked by reading, not compiled, not run on hardware
//
#include <hip/hip_runtime.h>
#include <hip/hip_bf16.h>

typedef __attribute__((ext_vector_type(16))) _Float16 v16h;
typedef __attribute__((ext_vector_type(8)))  _Float16 v8h;
typedef __attribute__((ext_vector_type(8)))  float    v8f;

#define EPS 1e-5f

// ---------------------------------------------------------------------------
// Elementwise helpers
// ---------------------------------------------------------------------------
__global__ void zero_f32(float* __restrict__ p, long n) {
    long i = (long)blockIdx.x * blockDim.x + threadIdx.x;
    if (i < n) p[i] = 0.0f;
}

__global__ void convert_f32_to_f16(const float* __restrict__ x, _Float16* __restrict__ y, long n) {
    long i = (long)blockIdx.x * blockDim.x + threadIdx.x;
    if (i < n) y[i] = (_Float16)x[i];
}

// w: [K][CI][CO] f32  ->  wt: [K][CO][CI] f16   (B pre-transpose for WMMA)
__global__ void convert_w_transpose(const float* __restrict__ w, _Float16* __restrict__ wt,
                                    int K, int CI, int CO) {
    long i = (long)blockIdx.x * blockDim.x + threadIdx.x;
    long total = (long)K * CI * CO;
    if (i >= total) return;
    int co = (int)(i % CO);
    long t = i / CO;
    int ci = (int)(t % CI);
    int k  = (int)(t / CI);
    wt[((long)k * CO + co) * CI + ci] = (_Float16)w[i];
}

// ---------------------------------------------------------------------------
// BN channel reduction: per-channel sum & sumsq over rows of x [nrows][64]
// ---------------------------------------------------------------------------
__global__ __launch_bounds__(256) void bn_reduce(const float* __restrict__ x, int nrows,
                                                 float* __restrict__ sums,
                                                 float* __restrict__ sumsq) {
    __shared__ float sm[256];
    __shared__ float sq[256];
    const int c = threadIdx.x & 63;
    const int g = threadIdx.x >> 6;   // 4 row-groups per block
    float s = 0.0f, q = 0.0f;
    for (long r = (long)blockIdx.x * 4 + g; r < nrows; r += (long)gridDim.x * 4) {
        float v = x[r * 64 + c];
        s += v;
        q += v * v;
    }
    sm[threadIdx.x] = s;
    sq[threadIdx.x] = q;
    __syncthreads();
    if (threadIdx.x < 64) {
        float st = sm[threadIdx.x] + sm[threadIdx.x + 64] + sm[threadIdx.x + 128] + sm[threadIdx.x + 192];
        float qt = sq[threadIdx.x] + sq[threadIdx.x + 64] + sq[threadIdx.x + 128] + sq[threadIdx.x + 192];
        unsafeAtomicAdd(&sums[threadIdx.x], st);
        unsafeAtomicAdd(&sumsq[threadIdx.x], qt);
    }
}

// y = f16( relu( bn(x) ) ), channel = i & 63
__global__ void bn_relu_to_f16(const float* __restrict__ x,
                               const float* __restrict__ sums, const float* __restrict__ sumsq,
                               const float* __restrict__ gamma, const float* __restrict__ beta,
                               _Float16* __restrict__ y, long n, float invN) {
    long i = (long)blockIdx.x * blockDim.x + threadIdx.x;
    if (i >= n) return;
    int c = (int)(i & 63);
    float mean = sums[c] * invN;
    float var  = sumsq[c] * invN - mean * mean;
    float v = (x[i] - mean) * rsqrtf(var + EPS) * gamma[c] + beta[c];
    y[i] = (_Float16)fmaxf(v, 0.0f);
}

// out = relu(out + bn(sc))
__global__ void final_add_bn_relu(float* __restrict__ out, const float* __restrict__ sc,
                                  const float* __restrict__ sums, const float* __restrict__ sumsq,
                                  const float* __restrict__ gamma, const float* __restrict__ beta,
                                  long n, float invN) {
    long i = (long)blockIdx.x * blockDim.x + threadIdx.x;
    if (i >= n) return;
    int c = (int)(i & 63);
    float mean = sums[c] * invN;
    float var  = sumsq[c] * invN - mean * mean;
    float b = (sc[i] - mean) * rsqrtf(var + EPS) * gamma[c] + beta[c];
    out[i] = fmaxf(out[i] + b, 0.0f);
}

// ---------------------------------------------------------------------------
// Sparse conv via WMMA: gather 16 rows -> [16 x CI] @ [CI x 64] -> atomic scatter
// Block = 256 threads = 8 waves. blockIdx.y = offset k. One 16-row tile / wave.
// Wt: [K][64][CI] f16 (pre-transposed), staged in LDS.
// ---------------------------------------------------------------------------
template <int CI>
__global__ __launch_bounds__(256) void sparse_conv_wmma(
    const _Float16* __restrict__ feats16,   // [N][CI]
    const _Float16* __restrict__ Wt,        // [K][64][CI]
    const int* __restrict__ in_idx,         // [K][M]
    const int* __restrict__ out_idx,        // [K][M]
    float* __restrict__ outacc,             // [N][64] (atomic accumulate)
    int M, int tilesPerK)
{
    __shared__ __align__(16) _Float16 w_lds[64 * CI];
    const int k = blockIdx.y;

    // cooperative LDS load of Wt[k] (16B per thread per step)
    const int chunks = (64 * CI) / 8;
    for (int c = threadIdx.x; c < chunks; c += blockDim.x)
        *(v8h*)&w_lds[c * 8] = *(const v8h*)&Wt[(long)k * (64 * CI) + c * 8];
    __syncthreads();

    const int wave = threadIdx.x >> 5;
    const int lane = threadIdx.x & 31;
    const int tile = blockIdx.x * 8 + wave;
    if (tile >= tilesPerK) return;

    const int mbase = tile * 16;
    const int r  = lane & 15;   // row (for A gather) / column (for B, D)
    const int hi = lane >> 4;

    const int inrow = in_idx[(long)k * M + mbase + r];
    const int myout = out_idx[(long)k * M + mbase + r];

    v8f acc[4] = {};
    const _Float16* arow = feats16 + (long)inrow * CI;

#pragma unroll
    for (int kk = 0; kk < CI; kk += 32) {
        // A fragment: lane lo holds K in [kk,kk+8)+[kk+16,kk+24); lane hi shifted by 8
        v8h a0 = *(const v8h*)(arow + kk + hi * 8);
        v8h a1 = *(const v8h*)(arow + kk + 16 + hi * 8);
        v16h a;
#pragma unroll
        for (int j = 0; j < 8; ++j) { a[j] = a0[j]; a[j + 8] = a1[j]; }

#pragma unroll
        for (int nt = 0; nt < 4; ++nt) {
            // B fragment: column nt*16+r, K in [kk+8*hi,+8)+[kk+16+8*hi,+8) (matches A)
            const _Float16* bp = &w_lds[(nt * 16 + r) * CI + kk + hi * 8];
            v8h b0 = *(const v8h*)bp;
            v8h b1 = *(const v8h*)(bp + 16);
            v16h b;
#pragma unroll
            for (int j = 0; j < 8; ++j) { b[j] = b0[j]; b[j + 8] = b1[j]; }

            acc[nt] = __builtin_amdgcn_wmma_f32_16x16x32_f16(
                false, a, false, b, (short)0, acc[nt], false, false);
        }
    }

    // scatter-add: lane owns column (lane&15); VGPR v = row 8*hi + v
    const int rbase = hi * 8;
#pragma unroll
    for (int nt = 0; nt < 4; ++nt) {
#pragma unroll
        for (int v = 0; v < 8; ++v) {
            int grow = __shfl(myout, rbase + v, 32);
            unsafeAtomicAdd(&outacc[(long)grow * 64 + nt * 16 + r], acc[nt][v]);
        }
    }
}

// ---------------------------------------------------------------------------
// Dense shortcut GEMM: sc[N][64] = feats16[N][32] @ Wsc[32][64] (Wt: [64][32])
// ---------------------------------------------------------------------------
__global__ __launch_bounds__(256) void dense_mm_wmma(
    const _Float16* __restrict__ feats16,   // [N][32]
    const _Float16* __restrict__ Wt,        // [64][32]
    float* __restrict__ sc,                 // [N][64]
    int rowTiles)
{
    __shared__ __align__(16) _Float16 w_lds[64 * 32];
    for (int c = threadIdx.x; c < (64 * 32) / 8; c += blockDim.x)
        *(v8h*)&w_lds[c * 8] = *(const v8h*)&Wt[c * 8];
    __syncthreads();

    const int wave = threadIdx.x >> 5;
    const int lane = threadIdx.x & 31;
    const int tile = blockIdx.x * 8 + wave;
    if (tile >= rowTiles) return;

    const int mbase = tile * 16;
    const int r  = lane & 15;
    const int hi = lane >> 4;

    const _Float16* arow = feats16 + (long)(mbase + r) * 32;
    v8h a0 = *(const v8h*)(arow + hi * 8);
    v8h a1 = *(const v8h*)(arow + 16 + hi * 8);
    v16h a;
#pragma unroll
    for (int j = 0; j < 8; ++j) { a[j] = a0[j]; a[j + 8] = a1[j]; }

    v8f acc[4] = {};
#pragma unroll
    for (int nt = 0; nt < 4; ++nt) {
        const _Float16* bp = &w_lds[(nt * 16 + r) * 32 + hi * 8];
        v8h b0 = *(const v8h*)bp;
        v8h b1 = *(const v8h*)(bp + 16);
        v16h b;
#pragma unroll
        for (int j = 0; j < 8; ++j) { b[j] = b0[j]; b[j + 8] = b1[j]; }
        acc[nt] = __builtin_amdgcn_wmma_f32_16x16x32_f16(
            false, a, false, b, (short)0, acc[nt], false, false);
    }

    const int rbase = hi * 8;
#pragma unroll
    for (int nt = 0; nt < 4; ++nt)
#pragma unroll
        for (int v = 0; v < 8; ++v)
            sc[(long)(mbase + rbase + v) * 64 + nt * 16 + r] = acc[nt][v];
}

// ---------------------------------------------------------------------------
// Launcher
// ---------------------------------------------------------------------------
extern "C" void kernel_launch(void* const* d_in, const int* in_sizes, int n_in,
                              void* d_out, int out_size, void* d_ws, size_t ws_size,
                              hipStream_t stream) {
    const float* feats    = (const float*)d_in[0];   // [N][32]
    const float* W1       = (const float*)d_in[1];   // [K][32][64]
    const float* gamma1   = (const float*)d_in[2];
    const float* beta1    = (const float*)d_in[3];
    const float* W2       = (const float*)d_in[4];   // [K][64][64]
    const float* Wsc      = (const float*)d_in[5];   // [32][64]
    const float* gamma_sc = (const float*)d_in[6];
    const float* beta_sc  = (const float*)d_in[7];
    const int*   in_idx   = (const int*)d_in[8];     // [K][M]
    const int*   out_idx  = (const int*)d_in[9];     // [K][M]
    float* out = (float*)d_out;                      // [N][64]

    const int CI = 32, CO = 64;
    const int N = in_sizes[0] / CI;
    const int K = in_sizes[1] / (CI * CO);
    const int M = in_sizes[8] / K;

    // workspace carve-out (256B aligned)
    size_t off = 0;
    auto carve = [&](size_t bytes) -> char* {
        char* p = (char*)d_ws + off;
        off += (bytes + 255) & ~(size_t)255;
        return p;
    };
    _Float16* feats16 = (_Float16*)carve((size_t)N * CI * sizeof(_Float16));
    _Float16* h16     = (_Float16*)carve((size_t)N * CO * sizeof(_Float16));
    float*    tmp     = (float*)   carve((size_t)N * CO * sizeof(float));  // conv1 acc, later shortcut
    _Float16* W1t     = (_Float16*)carve((size_t)K * CO * CI * sizeof(_Float16));
    _Float16* W2t     = (_Float16*)carve((size_t)K * CO * CO * sizeof(_Float16));
    _Float16* Wsct    = (_Float16*)carve((size_t)CO * CI * sizeof(_Float16));
    float*    stats   = (float*)   carve(4 * 64 * sizeof(float));
    float* sums1 = stats, *sumsq1 = stats + 64, *sums_s = stats + 128, *sumsq_s = stats + 192;
    (void)ws_size; (void)n_in; (void)out_size;

    const long nNC = (long)N * CO;        // N*64
    const float invN = 1.0f / (float)N;
    const int tilesPerK = M / 16;         // M = 200000 -> 12500
    const int rowTilesN = N / 16;         // N = 400000 -> 25000

    auto blocks1d = [](long n) { return (unsigned)((n + 255) / 256); };

    // 1) precision/layout conversions
    convert_f32_to_f16<<<blocks1d((long)N * CI), 256, 0, stream>>>(feats, feats16, (long)N * CI);
    convert_w_transpose<<<blocks1d((long)K * CI * CO), 256, 0, stream>>>(W1, W1t, K, CI, CO);
    convert_w_transpose<<<blocks1d((long)K * CO * CO), 256, 0, stream>>>(W2, W2t, K, CO, CO);
    convert_w_transpose<<<blocks1d((long)CI * CO), 256, 0, stream>>>(Wsc, Wsct, 1, CI, CO);

    // 2) zero accumulators + stats
    zero_f32<<<blocks1d(nNC), 256, 0, stream>>>(tmp, nNC);
    zero_f32<<<blocks1d(nNC), 256, 0, stream>>>(out, nNC);
    zero_f32<<<1, 256, 0, stream>>>(stats, 256);

    // 3) conv1: feats16 -> tmp (atomic scatter)
    {
        dim3 grid((tilesPerK + 7) / 8, K);
        sparse_conv_wmma<32><<<grid, 256, 0, stream>>>(feats16, W1t, in_idx, out_idx, tmp, M, tilesPerK);
    }

    // 4) BN1 stats + apply + ReLU -> h16
    bn_reduce<<<512, 256, 0, stream>>>(tmp, N, sums1, sumsq1);
    bn_relu_to_f16<<<blocks1d(nNC), 256, 0, stream>>>(tmp, sums1, sumsq1, gamma1, beta1, h16, nNC, invN);

    // 5) conv2: h16 -> out (atomic scatter)
    {
        dim3 grid((tilesPerK + 7) / 8, K);
        sparse_conv_wmma<64><<<grid, 256, 0, stream>>>(h16, W2t, in_idx, out_idx, out, M, tilesPerK);
    }

    // 6) shortcut GEMM -> tmp (reused), BN stats
    dense_mm_wmma<<<(rowTilesN + 7) / 8, 256, 0, stream>>>(feats16, Wsct, tmp, rowTilesN);
    bn_reduce<<<512, 256, 0, stream>>>(tmp, N, sums_s, sumsq_s);

    // 7) out = relu(out + bn(tmp))
    final_add_bn_relu<<<blocks1d(nNC), 256, 0, stream>>>(out, tmp, sums_s, sumsq_s,
                                                         gamma_sc, beta_sc, nNC, invN);
}
